// MIM_3839700763163
// MI455X (gfx1250) — matrix operand order repaired and checked by reading, compiled, not run on hardware
//
#include <hip/hip_runtime.h>
#include <hip/hip_bf16.h>

// ---------------------------------------------------------------------------
// Types for CDNA5 WMMA (wave32): A/B = 16 halves (8 VGPRs), C/D = 8 floats.
// ---------------------------------------------------------------------------
typedef __attribute__((ext_vector_type(16))) _Float16 v16h;
typedef __attribute__((ext_vector_type(8)))  float    v8f;
typedef __attribute__((ext_vector_type(4)))  unsigned int v4u;

union Frag { v16h h; v4u q[2]; unsigned int u[8]; };

// ---------------------------------------------------------------------------
// Implicit-GEMM conv (KS=5 pad=2, or KS=1 pad=0) on 32x32 maps, B=8.
//   in : [B, Cin, 32, 32] fp32
//   wt : [Cout, Cin, KS, KS] fp32
//   out: [B, Cout, 32, 32] fp32
// GEMM view: O[m,n] = sum_k W[m,k]*I[k,n], m=co, n=b*1024+y*32+x,
// K factored as (ky,kx) taps x Cin chunks of 32 (zero-padded if short).
// Block: 256 thr = 8 waves; tile 64(M) x 256(N) (one image, 8 rows x 32 cols).
// Wave tile: 32(M) x 64(N) = 2x4 WMMA accumulators.
// Weights staged one ky-row (KS taps) at a time -> 40 WMMAs per barrier pair.
// k-dim padded to 40 halves: 80B rows are 16B-aligned (ds_load_b128) and give
// a 20-bank stride whose 16 lane start-banks are disjoint 4-bank groups.
// ---------------------------------------------------------------------------
template<int KS>
__global__ void __launch_bounds__(256)
conv_wmma(const float* __restrict__ in, const float* __restrict__ wt,
          float* __restrict__ out, int Cin, int Cout)
{
    constexpr int PAD = KS / 2;
    constexpr int IR  = 8 + KS - 1;    // input rows staged (with halo)
    constexpr int IC  = 32 + KS - 1;   // input cols staged (with halo)
    constexpr int KP  = 40;            // padded k-dim (80B rows, 16B aligned)

    __shared__ _Float16 lds_in[IR][IC][KP];    // [row'][col'][kk]
    __shared__ _Float16 lds_w [KS][64][KP];    // [kx][m_local][kk]

    const int nb   = blockIdx.x;        // 0..31
    const int b    = nb >> 2;           // image
    const int y0   = (nb & 3) * 8;      // first output row of tile
    const int m0   = blockIdx.y * 64;   // first output channel of tile

    const int tid  = threadIdx.x;
    const int wv   = tid >> 5;
    const int lane = tid & 31;
    const int mW   = (wv >> 2) * 32;    // wave M offset (0 or 32)
    const int nW   = (wv & 3) * 64;     // wave N offset (0..192)
    const int lm   = lane & 15;
    const int kh   = (lane >> 4) & 1;

    v8f acc[2][4] = {};

    const int nch = (Cin + 31) >> 5;
    for (int cchunk = 0; cchunk < nch; ++cchunk) {
        const int cbase = cchunk << 5;
        const int kv    = (Cin - cbase) < 32 ? (Cin - cbase) : 32;

        __syncthreads();   // protect lds_in from previous chunk's readers

        // ---- stage input chunk (fp32 -> fp16, zero halo / zero k-pad) ----
        constexpr int TOTI = IR * IC * 32;
        for (int e = tid; e < TOTI; e += 256) {
            const int kk = e / (IR * IC);
            const int rm = e - kk * (IR * IC);
            const int rp = rm / IC;            // row' in tile
            const int cp = rm - rp * IC;       // col' in tile
            const int y  = y0 + rp - PAD;
            const int x  = cp - PAD;
            float v = 0.0f;
            if (kk < kv && (unsigned)y < 32u && (unsigned)x < 32u)
                v = in[(((b * Cin + cbase + kk) << 10) + (y << 5) + x)];
            lds_in[rp][cp][kk] = (_Float16)v;
        }

        // hint: pull next chunk's activations toward the WGP while we compute
        if (cchunk + 1 < nch) {
            const float* nx = in + (((size_t)(b * Cin + cbase + 32)) << 10);
            __builtin_prefetch(nx + (size_t)tid * 128, 0, 1);
        }

        for (int ky = 0; ky < KS; ++ky) {
            __syncthreads();  // prior ky-row's lds_w readers done
            // ---- stage weight row W[m0:m0+64, cbase:+32, ky, 0:KS] ----
            for (int e = tid; e < KS * 64 * 32; e += 256) {
                const int kk = e & 31;
                const int m  = (e >> 5) & 63;
                const int kx = e >> 11;
                float v = 0.0f;
                if (kk < kv && (m0 + m) < Cout)
                    v = wt[(((m0 + m) * Cin + cbase + kk) * (KS * KS))
                           + ky * KS + kx];
                lds_w[kx][m][kk] = (_Float16)v;
            }
            __syncthreads();

#pragma unroll
            for (int kx = 0; kx < KS; ++kx) {
                // ---- A fragments: 16x32 f16 layout; halves are contiguous
                // per lane-half -> two ds_load_b128 per fragment.
                Frag a[2];
#pragma unroll
                for (int mt = 0; mt < 2; ++mt) {
                    const _Float16* wrow = &lds_w[kx][mW + mt * 16 + lm][0];
                    a[mt].q[0] = *(const v4u*)(wrow + kh * 8);        // K lo half
                    a[mt].q[1] = *(const v4u*)(wrow + 16 + kh * 8);   // K hi half
                }
                // ---- B fragments: 32x16 f16; same K order as A (K lo half
                // 8*kh.., K hi half 16+8*kh..) -> two ds_load_b128 per fragment.
                Frag bf[4];
#pragma unroll
                for (int nt = 0; nt < 4; ++nt) {
                    const int nl = nW + nt * 16 + lm;        // 0..255
                    const _Float16* irow =
                        &lds_in[(nl >> 5) + ky][(nl & 31) + kx][0];
                    bf[nt].q[0] = *(const v4u*)(irow + kh * 8);
                    bf[nt].q[1] = *(const v4u*)(irow + 16 + kh * 8);
                }
                // ---- 8 WMMAs: D = A x B + C ----------------------------
#pragma unroll
                for (int mt = 0; mt < 2; ++mt)
#pragma unroll
                for (int nt = 0; nt < 4; ++nt)
                    acc[mt][nt] = __builtin_amdgcn_wmma_f32_16x16x32_f16(
                        false, a[mt].h, false, bf[nt].h,
                        (short)0, acc[mt][nt], false, false);
            }
        }
    }

    // ---- store: C/D layout — lane 0-15: N=lane, VGPR r: M=r; lanes 16-31 M=r+8
#pragma unroll
    for (int mt = 0; mt < 2; ++mt)
#pragma unroll
    for (int nt = 0; nt < 4; ++nt) {
        const int nl = nW + nt * 16 + lm;
        const int y  = y0 + (nl >> 5);
        const int x  = nl & 31;
#pragma unroll
        for (int r = 0; r < 8; ++r) {
            const int m = m0 + mW + mt * 16 + ((lane >> 4) << 3) + r;
            if (m < Cout)
                out[((b * Cout + m) << 10) + (y << 5) + x] = acc[mt][nt][r];
        }
    }
}

// ---------------------------------------------------------------------------
// Pointwise kernels. All state/gate tensors are [8, C, 32, 32] fp32.
// idx decomposition for C=64 tensors: b = idx>>16, ch = (idx>>10)&63, sp low 10.
// ---------------------------------------------------------------------------
__device__ __forceinline__ float sg(float x) { return 1.0f / (1.0f + __expf(-x)); }

#define EW_PRE                                              \
    const int idx = blockIdx.x * 256 + threadIdx.x;         \
    const int b  = idx >> 16;                               \
    const int ch = (idx >> 10) & 63;                        \
    const int sp = idx & 1023;
#define AT(p, C, c0) p[(((b) * (C) + (c0) + ch) << 10) + sp]

// st_cell part 1: c_new, m_new (in place), mem2 = concat(c_new, m_new)
__global__ void ew_stcell1(const float* __restrict__ xc, const float* __restrict__ hc,
                           const float* __restrict__ mc, float* __restrict__ c,
                           float* __restrict__ m, float* __restrict__ mem2)
{
    EW_PRE
    const float i_x = AT(xc, 448, 0),   f_x  = AT(xc, 448, 64),  g_x  = AT(xc, 448, 128);
    const float i_xp= AT(xc, 448, 192), f_xp = AT(xc, 448, 256), g_xp = AT(xc, 448, 320);
    const float i_h = AT(hc, 256, 0),   f_h  = AT(hc, 256, 64),  g_h  = AT(hc, 256, 128);
    const float i_m = AT(mc, 192, 0),   f_m  = AT(mc, 192, 64),  g_m  = AT(mc, 192, 128);
    const float cv = c[idx], mv = m[idx];
    const float cn = sg(f_x + f_h + 1.0f) * cv + sg(i_x + i_h) * tanhf(g_x + g_h);
    const float mn = sg(f_xp + f_m + 1.0f) * mv + sg(i_xp + i_m) * tanhf(g_xp + g_m);
    c[idx] = cn; m[idx] = mn;
    mem2[((b * 128 + ch) << 10) + sp]      = cn;
    mem2[((b * 128 + 64 + ch) << 10) + sp] = mn;
}

// st_cell part 2: h = sig(o_x + o_h + oconv) * tanh(lconv)
__global__ void ew_stcell2(const float* __restrict__ xc, const float* __restrict__ hc,
                           const float* __restrict__ oconv, const float* __restrict__ lconv,
                           float* __restrict__ hout)
{
    EW_PRE
    const float o_x = AT(xc, 448, 384), o_h = AT(hc, 256, 192);
    hout[idx] = sg(o_x + o_h + oconv[idx]) * tanhf(lconv[idx]);
}

// gated (MIMN / internal MIMS): split order i,g,f,o
__global__ void ew_gated(const float* __restrict__ hc, const float* __restrict__ xc,
                         const float* __restrict__ ctw, const float* __restrict__ ocw,
                         float* __restrict__ c, float* __restrict__ hout)
{
    EW_PRE
    const float i_h = AT(hc, 256, 0), g_h = AT(hc, 256, 64);
    const float f_h = AT(hc, 256, 128), o_h = AT(hc, 256, 192);
    const float i_x = AT(xc, 256, 0), g_x = AT(xc, 256, 64);
    const float f_x = AT(xc, 256, 128), o_x = AT(xc, 256, 192);
    const float cv = c[idx];
    const int cw = (ch << 10) + sp;
    const float i_c = cv * ctw[cw];
    const float f_c = cv * ctw[(64 << 10) + cw];
    const float cn = sg(f_h + f_c + f_x + 1.0f) * cv + sg(i_h + i_c + i_x) * tanhf(g_h + g_x);
    hout[idx] = sg(o_h + o_x + cn * ocw[cw]) * tanhf(cn);
    c[idx] = cn;
}

// mimblock combine: new_m (in place on mem), new_c, cellcat
__global__ void ew_mbcomb(const float* __restrict__ sc, const float* __restrict__ tc,
                          const float* __restrict__ xb, float* __restrict__ m,
                          const float* __restrict__ c2, float* __restrict__ cout,
                          float* __restrict__ cellcat)
{
    EW_PRE
    const float i_s = AT(sc, 256, 0), g_s = AT(sc, 256, 64), f_s = AT(sc, 256, 128);
    const float i_t = AT(tc, 192, 0), g_t = AT(tc, 192, 64);
    const float i_x = AT(xb, 256, 0), g_x = AT(xb, 256, 64), f_x = AT(xb, 256, 128);
    const float mv = m[idx];
    const float nm = sg(f_x + f_s + 1.0f) * mv + sg(i_x + i_s) * tanhf(g_x + g_s);
    const float nc = c2[idx] + sg(i_x + i_t) * tanhf(g_x + g_t);
    m[idx] = nm; cout[idx] = nc;
    cellcat[((b * 128 + ch) << 10) + sp]      = nc;
    cellcat[((b * 128 + 64 + ch) << 10) + sp] = nm;
}

// mimblock final: h = sig(o_x + o_t + o_s) * tanh(conv1(cell))
__global__ void ew_mbfinal(const float* __restrict__ xb, const float* __restrict__ tc,
                           const float* __restrict__ sc, const float* __restrict__ lastc,
                           float* __restrict__ hout)
{
    EW_PRE
    const float o_x = AT(xb, 256, 192), o_t = AT(tc, 192, 128), o_s = AT(sc, 256, 192);
    hout[idx] = sg(o_x + o_t + o_s) * tanhf(lastc[idx]);
}

__global__ void ew_sub(const float* __restrict__ a, const float* __restrict__ bb,
                       float* __restrict__ o)
{
    const int idx = blockIdx.x * 256 + threadIdx.x;
    o[idx] = a[idx] - bb[idx];
}

__global__ void fill0(float* p, int n)
{
    const int idx = blockIdx.x * 256 + threadIdx.x;
    if (idx < n) p[idx] = 0.0f;
}

// frames0 [8,10,1,128,128] + frames1 [8,10,1,128,128] -> patches [20,8,16,32,32]
__global__ void patchify(const float* __restrict__ f0, const float* __restrict__ f1,
                         float* __restrict__ patches)
{
    const int idx = blockIdx.x * 256 + threadIdx.x;       // 20*8*16*1024
    const int sp = idx & 1023;
    const int x = sp & 31, y = sp >> 5;
    const int p = (idx >> 10) & 15;
    const int b = (idx >> 14) & 7;
    const int t = idx >> 17;
    const int py = p >> 2, px = p & 3;
    const float* src = (t < 10) ? f0 : f1;
    const int tt = (t < 10) ? t : (t - 10);
    patches[idx] = src[((b * 10 + tt) * 128 + (y * 4 + py)) * 128 + (x * 4 + px)];
}

// x_gen [8,16,32,32] -> d_out frame f of [8,10,1,128,128]
__global__ void unpatchify(const float* __restrict__ xg, float* __restrict__ out, int f)
{
    const int idx = blockIdx.x * 256 + threadIdx.x;       // 8*16*1024
    const int sp = idx & 1023;
    const int x = sp & 31, y = sp >> 5;
    const int p = (idx >> 10) & 15;
    const int b = idx >> 14;
    const int py = p >> 2, px = p & 3;
    out[((b * 10 + f) * 128 + (y * 4 + py)) * 128 + (x * 4 + px)] = xg[idx];
}

// ---------------------------------------------------------------------------
// Host orchestration
// ---------------------------------------------------------------------------
static inline void conv5(const float* in, const float* w, float* out,
                         int Cin, int Cout, hipStream_t s)
{
    hipLaunchKernelGGL((conv_wmma<5>), dim3(32, (Cout + 63) / 64), dim3(256), 0, s,
                       in, w, out, Cin, Cout);
}
static inline void conv1(const float* in, const float* w, float* out,
                         int Cin, int Cout, hipStream_t s)
{
    hipLaunchKernelGGL((conv_wmma<1>), dim3(32, (Cout + 63) / 64), dim3(256), 0, s,
                       in, w, out, Cin, Cout);
}

extern "C" void kernel_launch(void* const* d_in, const int* in_sizes, int n_in,
                              void* d_out, int out_size, void* d_ws, size_t ws_size,
                              hipStream_t stream)
{
    (void)in_sizes; (void)n_in; (void)out_size; (void)ws_size;
    const float* frames0 = (const float*)d_in[0];
    const float* frames1 = (const float*)d_in[1];
    const float* st_cx   = (const float*)d_in[2];
    const float* st_ch   = (const float*)d_in[3];
    const float* st_cm   = (const float*)d_in[4];
    const float* st_co   = (const float*)d_in[5];
    const float* st_cl   = (const float*)d_in[6];
    const float* mb_t    = (const float*)d_in[7];
    const float* mb_s    = (const float*)d_in[8];
    const float* mb_x    = (const float*)d_in[9];
    const float* mb_mh   = (const float*)d_in[10];
    const float* mb_mx   = (const float*)d_in[11];
    const float* mb_ctw  = (const float*)d_in[12];
    const float* mb_ocw  = (const float*)d_in[13];
    const float* mb_last = (const float*)d_in[14];
    const float* mn_ch   = (const float*)d_in[15];
    const float* mn_cx   = (const float*)d_in[16];
    const float* mn_ctw  = (const float*)d_in[17];
    const float* mn_ocw  = (const float*)d_in[18];
    const float* w_last  = (const float*)d_in[19];

    // weight strides per layer index i (elements)
    const size_t S_MB_T = 192 * 64 * 25, S_MB4 = 256 * 64 * 25;
    const size_t S_CTW = 128 * 1024, S_OCW = 64 * 1024, S_MB_L = 64 * 128;

    float* ws = (float*)d_ws;
    size_t off = 0;
    auto alloc = [&](size_t n) { float* p = ws + off; off += n; return p; };
    const size_t S64 = 8 * 64 * 1024, S16 = 8 * 16 * 1024;

    // ---- recurrent state (zeroed every call; deterministic) ----
    float* stateBase = ws;
    float* h0a = alloc(S64); float* h0b = alloc(S64);
    float* hsL[4]; hsL[1] = alloc(S64); hsL[2] = alloc(S64); hsL[3] = alloc(S64);
    float* csL[4]; for (int i = 0; i < 4; ++i) csL[i] = alloc(S64);
    float* mem = alloc(S64);
    float* dh[3]; float* dc[3]; float* cc3[3];
    for (int i = 0; i < 3; ++i) dh[i] = alloc(S64);
    for (int i = 0; i < 3; ++i) dc[i] = alloc(S64);
    for (int i = 0; i < 3; ++i) cc3[i] = alloc(S64);
    float* xgen = alloc(S16);
    float* zbuf = alloc(S64);          // stays zero: diff_h at t=0
    float* diff = alloc(S64);
    const size_t stateCount = off;

    // ---- temporaries ----
    float* patches = alloc(20 * S16);
    float* xc   = alloc((size_t)8 * 448 * 1024);
    float* hc   = alloc((size_t)8 * 256 * 1024);
    float* mc   = alloc((size_t)8 * 192 * 1024);
    float* mem2 = alloc((size_t)8 * 128 * 1024);
    float* oconv = alloc(S64);
    float* lconv = alloc(S64);
    float* sconv = alloc((size_t)8 * 256 * 1024);
    float* tconv = alloc((size_t)8 * 192 * 1024);
    float* xbv   = alloc((size_t)8 * 256 * 1024);
    float* gA    = alloc((size_t)8 * 256 * 1024);
    float* gB    = alloc((size_t)8 * 256 * 1024);
    float* c2    = alloc(S64);
    float* cellcat = alloc((size_t)8 * 128 * 1024);
    float* lastc = alloc(S64);

    const dim3 B256(256);
    const dim3 G64((unsigned)(S64 / 256));

    hipLaunchKernelGGL(fill0, dim3((unsigned)((stateCount + 255) / 256)), B256, 0,
                       stream, stateBase, (int)stateCount);
    hipLaunchKernelGGL(patchify, dim3((unsigned)(20 * S16 / 256)), B256, 0,
                       stream, frames0, frames1, patches);

    float* h0_cur = h0a; float* h0_nxt = h0b;

    for (int t = 0; t < 19; ++t) {
        const float* net = (t < 10) ? (patches + (size_t)t * S16) : xgen;

        // ---------------- SpatioTemporal LSTM (layer 0) ----------------
        conv5(net, st_cx, xc, 16, 448, stream);
        conv5(h0_cur, st_ch, hc, 64, 256, stream);
        conv5(mem, st_cm, mc, 64, 192, stream);
        hipLaunchKernelGGL(ew_stcell1, G64, B256, 0, stream, xc, hc, mc, csL[0], mem, mem2);
        conv5(mem2, st_co, oconv, 128, 64, stream);
        conv1(mem2, st_cl, lconv, 128, 64, stream);
        hipLaunchKernelGGL(ew_stcell2, G64, B256, 0, stream, xc, hc, oconv, lconv, h0_nxt);
        if (t >= 1)
            hipLaunchKernelGGL(ew_sub, G64, B256, 0, stream, h0_nxt, h0_cur, diff);
        { float* tmp = h0_cur; h0_cur = h0_nxt; h0_nxt = tmp; }

        // ---------------- layers 1..3: MIMN + MIMBlock ----------------
        for (int L = 1; L < 4; ++L) {
            const int i = L - 1;
            if (t >= 1) {
                const float* din = (i == 0) ? diff : dh[i - 1];
                conv5(dh[i], mn_ch + (size_t)i * S_MB4, gA, 64, 256, stream);
                conv5(din,   mn_cx + (size_t)i * S_MB4, gB, 64, 256, stream);
                hipLaunchKernelGGL(ew_gated, G64, B256, 0, stream, gA, gB,
                                   mn_ctw + (size_t)i * S_CTW,
                                   mn_ocw + (size_t)i * S_OCW, dc[i], dh[i]);
            }
            const float* dhi  = (t == 0) ? zbuf : dh[i];
            const float* xlow = (L == 1) ? h0_cur : hsL[L - 1];
            conv5(mem,     mb_s  + (size_t)i * S_MB4,  sconv, 64, 256, stream);
            conv5(hsL[L],  mb_t  + (size_t)i * S_MB_T, tconv, 64, 192, stream);
            conv5(xlow,    mb_x  + (size_t)i * S_MB4,  xbv,   64, 256, stream);
            conv5(csL[L],  mb_mh + (size_t)i * S_MB4,  gA,    64, 256, stream);
            conv5(dhi,     mb_mx + (size_t)i * S_MB4,  gB,    64, 256, stream);
            hipLaunchKernelGGL(ew_gated, G64, B256, 0, stream, gA, gB,
                               mb_ctw + (size_t)i * S_CTW,
                               mb_ocw + (size_t)i * S_OCW, cc3[i], c2);
            hipLaunchKernelGGL(ew_mbcomb, G64, B256, 0, stream,
                               sconv, tconv, xbv, mem, c2, csL[L], cellcat);
            conv1(cellcat, mb_last + (size_t)i * S_MB_L, lastc, 128, 64, stream);
            hipLaunchKernelGGL(ew_mbfinal, G64, B256, 0, stream,
                               xbv, tconv, sconv, lastc, hsL[L]);
        }

        // ---------------- x_gen + output frames ----------------
        conv1(hsL[3], w_last, xgen, 64, 16, stream);
        if (t >= 9)
            hipLaunchKernelGGL(unpatchify, dim3((unsigned)(S16 / 256)), B256, 0,
                               stream, xgen, (float*)d_out, t - 9);
    }
}
